// Attention_81716047774180
// MI455X (gfx1250) — matrix operand order repaired and checked by reading, not compiled, not run on hardware
//
#include <hip/hip_runtime.h>
#include <hip/hip_bf16.h>

// Problem constants (match reference)
#define BATCH   64
#define NTOK    320          // 8*8 + 16*16
#define DIMC    768
#define NHEAD   12
#define DHEAD   64
#define MGLOB   (BATCH * NTOK)      // 20480 rows
#define QKVC    (3 * DIMC)          // 2304 cols

typedef __attribute__((ext_vector_type(16))) _Float16 v16h;
typedef __attribute__((ext_vector_type(8)))  _Float16 v8h;
typedef __attribute__((ext_vector_type(8)))  float    v8f;
typedef __attribute__((ext_vector_type(4)))  int      v4i;

#if defined(__has_builtin)
#if __has_builtin(__builtin_amdgcn_global_load_async_to_lds_b128) && \
    __has_builtin(__builtin_amdgcn_s_wait_asynccnt)
#define USE_ASYNC_LDS 1
#endif
#endif

#if USE_ASYNC_LDS
typedef __attribute__((address_space(1))) v4i gv4i;   // global 16B vector
typedef __attribute__((address_space(3))) v4i lv4i;   // LDS    16B vector
#endif

// ---------------------------------------------------------------------------
// WMMA fragment helpers (layouts per CDNA5 ISA 7.12.2, wave32)
// A 16x32 f16: lane L holds row M=L&15; halves 0..7 -> K = i + 8*(L/16),
//              halves 8..15 -> K = 16 + (i-8) + 8*(L/16)
// B 32x16 f16: lane L holds col N=L&15; same K mapping as A
// C/D 16x16 f32: c[r] = C[r + 8*(L/16)][L&15]
// ---------------------------------------------------------------------------
__device__ __forceinline__ v16h ldA(const _Float16* row, int kk, int lane) {
    int g8 = (lane >> 4) << 3;
    v8h lo = *(const v8h*)(row + kk + g8);
    v8h hi = *(const v8h*)(row + kk + 16 + g8);
    v16h r;
#pragma unroll
    for (int i = 0; i < 8; ++i) { r[i] = lo[i]; r[i + 8] = hi[i]; }
    return r;
}
__device__ __forceinline__ v16h ldB(const _Float16* col, int kk, int lane) {
    int g8 = (lane >> 4) << 3;
    v8h lo = *(const v8h*)(col + kk + g8);
    v8h hi = *(const v8h*)(col + kk + 16 + g8);
    v16h r;
#pragma unroll
    for (int i = 0; i < 8; ++i) { r[i] = lo[i]; r[i + 8] = hi[i]; }
    return r;
}
__device__ __forceinline__ v8f wmma_f16(v16h a, v16h b, v8f c) {
    return __builtin_amdgcn_wmma_f32_16x16x32_f16(false, a, false, b,
                                                  (short)0, c, false, false);
}

// ---------------------------------------------------------------------------
// Prep kernels
// ---------------------------------------------------------------------------
__global__ void k_cvt_f16(const float* __restrict__ src,
                          _Float16* __restrict__ dst, int n) {
    int i = blockIdx.x * blockDim.x + threadIdx.x;
    if (i < n) dst[i] = (_Float16)src[i];
}

// wt[n*K + k] = (f16) w[k*N + n]   (transpose + convert)
__global__ void k_transpose_f16(const float* __restrict__ w,
                                _Float16* __restrict__ wt, int K, int N) {
    int i = blockIdx.x * blockDim.x + threadIdx.x;
    if (i >= K * N) return;
    int n = i / K, k = i % K;
    wt[n * K + k] = (_Float16)w[k * N + n];
}

// biasM[h][i][j] = rpb[h*n_bias + rel[i*N+j]]
__global__ void k_bias(const float* __restrict__ rpb,
                       const int* __restrict__ rel,
                       float* __restrict__ biasM, int n_bias) {
    int i = blockIdx.x * blockDim.x + threadIdx.x;
    if (i >= NHEAD * NTOK * NTOK) return;
    int h = i / (NTOK * NTOK), r = i % (NTOK * NTOK);
    biasM[i] = rpb[h * n_bias + rel[r]];
}

// ---------------------------------------------------------------------------
// Staged GEMM building blocks: 256-thread WG computes a 128x128 tile.
// K is consumed in 32-half slices staged in LDS (double buffered, async).
// Each wave (8 per WG, arranged 4(M) x 2(N)) computes 32x64 -> 8 WMMAs/step.
// ---------------------------------------------------------------------------
#define KC      32
#define LDS_STR 40                  // KC + 8-half pad (80B rows, bank spread)
#define NKS     (DIMC / KC)         // 24 K-steps

// copy 128 rows x 32 halves (row-major, gstride halves) into LDS tile
__device__ __forceinline__ void stage32(const _Float16* g, int gstride,
                                        _Float16* l, int t) {
#pragma unroll
    for (int i = 0; i < 2; ++i) {
        int seg = t + i * 256;          // 512 x 16B segments
        int row = seg >> 2;
        int cs  = (seg & 3) << 3;
        const _Float16* s = g + (size_t)row * gstride + cs;
        _Float16* d = l + row * LDS_STR + cs;
#if USE_ASYNC_LDS
        __builtin_amdgcn_global_load_async_to_lds_b128((gv4i*)s, (lv4i*)d, 0, 0);
#else
        *(v8h*)d = *(const v8h*)s;
#endif
    }
}

__device__ __forceinline__ void gemm_mainloop(const _Float16* __restrict__ Ag,
                                              const _Float16* __restrict__ Bg,
                                              _Float16 (&sA)[2][128 * LDS_STR],
                                              _Float16 (&sB)[2][128 * LDS_STR],
                                              int t, int lane, int mw, int nw,
                                              v8f (&acc)[2][4]) {
    stage32(Ag, DIMC, sA[0], t);
    stage32(Bg, DIMC, sB[0], t);
    int cl = lane & 15;
#pragma unroll 1
    for (int k = 0; k < NKS; ++k) {
        int cur = k & 1;
        if (k + 1 < NKS) {
            stage32(Ag + (k + 1) * KC, DIMC, sA[1 - cur], t);
            stage32(Bg + (k + 1) * KC, DIMC, sB[1 - cur], t);
#if USE_ASYNC_LDS
            __builtin_amdgcn_s_wait_asynccnt(4);   // current slice landed
#endif
        } else {
#if USE_ASYNC_LDS
            __builtin_amdgcn_s_wait_asynccnt(0);
#endif
        }
        __syncthreads();
        v16h a0 = ldA(&sA[cur][(mw * 32 + cl) * LDS_STR], 0, lane);
        v16h a1 = ldA(&sA[cur][(mw * 32 + 16 + cl) * LDS_STR], 0, lane);
#pragma unroll
        for (int nt = 0; nt < 4; ++nt) {
            v16h bf = ldB(&sB[cur][(nw * 64 + nt * 16 + cl) * LDS_STR], 0, lane);
            acc[0][nt] = wmma_f16(a0, bf, acc[0][nt]);
            acc[1][nt] = wmma_f16(a1, bf, acc[1][nt]);
        }
        __syncthreads();
    }
}

// ---------------------------------------------------------------------------
// Kernel 1: QKV GEMM  (xh[20480x768] @ wqkvT -> q (scaled), k (f16), vT (f16))
// grid = (20480/128) x (2304/128) = 160 x 18 WGs of 256 threads
// ---------------------------------------------------------------------------
#define QKV_NB (QKVC / 128)         // 18
__global__ void __launch_bounds__(256, 1)
k_qkv_gemm(const _Float16* __restrict__ xh,
           const _Float16* __restrict__ wqkvT,
           const float* __restrict__ qkv_b,
           _Float16* __restrict__ qh,
           _Float16* __restrict__ kh,
           _Float16* __restrict__ vt) {
    __shared__ _Float16 sA[2][128 * LDS_STR];
    __shared__ _Float16 sB[2][128 * LDS_STR];
    int t = threadIdx.x, lane = t & 31, wv = t >> 5;
    int mw = wv >> 1, nw = wv & 1;
    int mblk = blockIdx.x / QKV_NB, nblk = blockIdx.x % QKV_NB;

    v8f acc[2][4] = {};
    gemm_mainloop(xh + (size_t)mblk * 128 * DIMC,
                  wqkvT + (size_t)nblk * 128 * DIMC, sA, sB,
                  t, lane, mw, nw, acc);

    // epilogue: bias + route.  64-col group per wave lies in one (s, head)
    int cbase = nblk * 128 + nw * 64;
    int s  = cbase / DIMC;                 // 0=q 1=k 2=v
    int hh = (cbase % DIMC) / DHEAD;
    int rbase = (lane >> 4) << 3;
    int cl = lane & 15;
#pragma unroll
    for (int at = 0; at < 2; ++at)
#pragma unroll
        for (int nt = 0; nt < 4; ++nt) {
            int col = cbase + nt * 16 + cl;
            int dd  = (col % DIMC) % DHEAD;
            float bias = qkv_b[col];
#pragma unroll
            for (int r = 0; r < 8; ++r) {
                int m  = mblk * 128 + mw * 32 + at * 16 + rbase + r;
                int b  = m / NTOK, n_ = m % NTOK;
                float v = acc[at][nt][r] + bias;
                size_t bh = (size_t)b * NHEAD + hh;
                if (s == 0)      qh[(bh * NTOK + n_) * DHEAD + dd] = (_Float16)(v * 0.125f);
                else if (s == 1) kh[(bh * NTOK + n_) * DHEAD + dd] = (_Float16)v;
                else             vt[(bh * DHEAD + dd) * NTOK + n_] = (_Float16)v;
            }
        }
}

// ---------------------------------------------------------------------------
// Kernel 2: attention.  one wave = (b, h, 16-query block); block = 2 waves
// ---------------------------------------------------------------------------
#define QBLK (NTOK / 16)            // 20 query blocks / (b,h)
#define KTIL (NTOK / 16)            // 20 key tiles
__global__ void k_attn(const _Float16* __restrict__ qh,
                       const _Float16* __restrict__ kh,
                       const _Float16* __restrict__ vt,
                       const float*    __restrict__ biasM,
                       _Float16* __restrict__ ao) {
    __shared__ float    sS[2][16 * NTOK];     // 40 KB scores (f32)
    __shared__ _Float16 sP[2][16 * NTOK];     // 20 KB probs (f16)
    __shared__ float    sMax[2][16];
    __shared__ float    sSum[2][16];

    int lane = threadIdx.x & 31;
    int wid  = threadIdx.x >> 5;
    int w    = blockIdx.x * 2 + wid;
    int bh   = w / QBLK;
    int qblk = w % QBLK;
    int b    = bh / NHEAD, h = bh % NHEAD;
    int rbase = (lane >> 4) << 3;
    int cl    = lane & 15;

    // q A-fragments (d=64 -> two K=32 chunks), pre-scaled in kernel 1
    const _Float16* qrow = qh + (size_t)(bh * NTOK + qblk * 16 + cl) * DHEAD;
    v16h aq0 = ldA(qrow, 0, lane);
    v16h aq1 = ldA(qrow, 32, lane);

    // ---- pass 1: scores + bias, write LDS, track row max --------------------
    float rmax[8];
#pragma unroll
    for (int r = 0; r < 8; ++r) rmax[r] = -3.4e38f;

    for (int j = 0; j < KTIL; ++j) {
        const _Float16* krow = kh + (size_t)(bh * NTOK + j * 16 + cl) * DHEAD;
        v8f acc = {};
        acc = wmma_f16(aq0, ldB(krow, 0, lane), acc);
        acc = wmma_f16(aq1, ldB(krow, 32, lane), acc);
        int col = j * 16 + cl;
#pragma unroll
        for (int r = 0; r < 8; ++r) {
            int lrow = rbase + r;
            int grow = qblk * 16 + lrow;
            float s = acc[r] + biasM[((size_t)h * NTOK + grow) * NTOK + col];
            sS[wid][lrow * NTOK + col] = s;
            rmax[r] = fmaxf(rmax[r], s);
        }
    }
    // reduce max across the 16 lanes that hold one tile-row
#pragma unroll
    for (int m = 8; m >= 1; m >>= 1)
#pragma unroll
        for (int r = 0; r < 8; ++r)
            rmax[r] = fmaxf(rmax[r], __shfl_xor(rmax[r], m, 16));
    if (cl == 0)
#pragma unroll
        for (int r = 0; r < 8; ++r) sMax[wid][rbase + r] = rmax[r];
    __syncthreads();

    // ---- pass 2: exp, P (f16) to LDS, row sums ------------------------------
    float rsum[8] = {};
#pragma unroll
    for (int r = 0; r < 8; ++r) {
        int lrow = rbase + r;
        float mx = sMax[wid][lrow];
        for (int j = 0; j < KTIL; ++j) {
            int c = j * 16 + cl;
            float e = __expf(sS[wid][lrow * NTOK + c] - mx);
            sP[wid][lrow * NTOK + c] = (_Float16)e;
            rsum[r] += e;
        }
    }
#pragma unroll
    for (int m = 8; m >= 1; m >>= 1)
#pragma unroll
        for (int r = 0; r < 8; ++r)
            rsum[r] += __shfl_xor(rsum[r], m, 16);
    if (cl == 0)
#pragma unroll
        for (int r = 0; r < 8; ++r) sSum[wid][rbase + r] = rsum[r];
    __syncthreads();

    // ---- pass 3: out = P @ V  (10 K-chunks of 32 keys, 4 d-tiles) -----------
    v8f co[4] = {};
    const _Float16* prow = &sP[wid][cl * NTOK];
    for (int kb = 0; kb < NTOK; kb += 32) {
        v16h pa = ldA(prow, kb, lane);
#pragma unroll
        for (int nt = 0; nt < 4; ++nt) {
            const _Float16* vcol = vt + (size_t)(bh * DHEAD + nt * 16 + cl) * NTOK;
            co[nt] = wmma_f16(pa, ldB(vcol, kb, lane), co[nt]);
        }
    }
    // epilogue: divide by row sum, write (B, N, H*d) f16 for proj GEMM
#pragma unroll
    for (int nt = 0; nt < 4; ++nt)
#pragma unroll
        for (int r = 0; r < 8; ++r) {
            int lrow = rbase + r;
            int grow = qblk * 16 + lrow;
            float v  = co[nt][r] / sSum[wid][lrow];
            ao[((size_t)b * NTOK + grow) * DIMC + h * DHEAD + nt * 16 + cl] =
                (_Float16)v;
        }
}

// ---------------------------------------------------------------------------
// Kernel 3: proj GEMM  (ao[20480x768] @ wprojT + proj_b -> out f32)
// grid = 160 x 6 WGs of 256 threads
// ---------------------------------------------------------------------------
#define PROJ_NB (DIMC / 128)        // 6
__global__ void __launch_bounds__(256, 1)
k_proj_gemm(const _Float16* __restrict__ ao,
            const _Float16* __restrict__ wprojT,
            const float* __restrict__ proj_b,
            float* __restrict__ out) {
    __shared__ _Float16 sA[2][128 * LDS_STR];
    __shared__ _Float16 sB[2][128 * LDS_STR];
    int t = threadIdx.x, lane = t & 31, wv = t >> 5;
    int mw = wv >> 1, nw = wv & 1;
    int mblk = blockIdx.x / PROJ_NB, nblk = blockIdx.x % PROJ_NB;

    v8f acc[2][4] = {};
    gemm_mainloop(ao + (size_t)mblk * 128 * DIMC,
                  wprojT + (size_t)nblk * 128 * DIMC, sA, sB,
                  t, lane, mw, nw, acc);

    int rbase = (lane >> 4) << 3;
    int cl = lane & 15;
#pragma unroll
    for (int at = 0; at < 2; ++at)
#pragma unroll
        for (int nt = 0; nt < 4; ++nt) {
            int col = nblk * 128 + nw * 64 + nt * 16 + cl;
            float bias = proj_b[col];
#pragma unroll
            for (int r = 0; r < 8; ++r) {
                int m = mblk * 128 + mw * 32 + at * 16 + rbase + r;
                out[(size_t)m * DIMC + col] = acc[at][nt][r] + bias;
            }
        }
}

// ---------------------------------------------------------------------------
extern "C" void kernel_launch(void* const* d_in, const int* in_sizes, int n_in,
                              void* d_out, int out_size, void* d_ws, size_t ws_size,
                              hipStream_t stream) {
    const float* x      = (const float*)d_in[0];
    const float* qkv_w  = (const float*)d_in[1];
    const float* qkv_b  = (const float*)d_in[2];
    const float* proj_w = (const float*)d_in[3];
    const float* proj_b = (const float*)d_in[4];
    const float* rpb    = (const float*)d_in[5];
    const int*   rel    = (const int*)d_in[6];
    int n_bias = in_sizes[5] / NHEAD;
    float* out = (float*)d_out;

    char* ws = (char*)d_ws;
    size_t o = 0;
    _Float16* xh     = (_Float16*)(ws + o); o += (size_t)MGLOB * DIMC * 2;   // 31.5 MB
    _Float16* wqkvT  = (_Float16*)(ws + o); o += (size_t)QKVC * DIMC * 2;    //  3.5 MB
    _Float16* wprojT = (_Float16*)(ws + o); o += (size_t)DIMC * DIMC * 2;    //  1.2 MB
    _Float16* qh     = (_Float16*)(ws + o); o += (size_t)MGLOB * DIMC * 2;   // 31.5 MB
    _Float16* kh     = (_Float16*)(ws + o); o += (size_t)MGLOB * DIMC * 2;   // 31.5 MB
    _Float16* vt     = (_Float16*)(ws + o); o += (size_t)MGLOB * DIMC * 2;   // 31.5 MB
    _Float16* ao     = (_Float16*)(ws + o); o += (size_t)MGLOB * DIMC * 2;   // 31.5 MB
    float*    biasM  = (float*)(ws + o);    o += (size_t)NHEAD * NTOK * NTOK * 4; // 4.9 MB

    // prep
    {
        int n = MGLOB * DIMC;
        k_cvt_f16<<<(n + 255) / 256, 256, 0, stream>>>(x, xh, n);
    }
    {
        int n = DIMC * QKVC;
        k_transpose_f16<<<(n + 255) / 256, 256, 0, stream>>>(qkv_w, wqkvT, DIMC, QKVC);
    }
    {
        int n = DIMC * DIMC;
        k_transpose_f16<<<(n + 255) / 256, 256, 0, stream>>>(proj_w, wprojT, DIMC, DIMC);
    }
    {
        int n = NHEAD * NTOK * NTOK;
        k_bias<<<(n + 255) / 256, 256, 0, stream>>>(rpb, rel, biasM, n_bias);
    }

    // QKV GEMM: 160 x 18 tiles of 128x128
    k_qkv_gemm<<<(MGLOB / 128) * QKV_NB, 256, 0, stream>>>(xh, wqkvT, qkv_b,
                                                           qh, kh, vt);
    // attention: 64*12*20 = 15360 waves, 2 waves/block
    k_attn<<<BATCH * NHEAD * QBLK / 2, 64, 0, stream>>>(qh, kh, vt, biasM, ao);

    // proj GEMM: 160 x 6 tiles of 128x128
    k_proj_gemm<<<(MGLOB / 128) * PROJ_NB, 256, 0, stream>>>(ao, wprojT,
                                                             proj_b, out);
    (void)n_in; (void)out_size; (void)ws_size;
}
